// LSTMAutoEncoder_53609781788852
// MI455X (gfx1250) — matrix operand order repaired and checked by reading, not compiled, not run on hardware
//
#include <hip/hip_runtime.h>

// Problem constants (from the reference): T=512, B=64, I=256, H=512, O=256.
#define T_STEPS 512
#define BB 64
#define II 256
#define HH 512
#define OO 256
#define G4 2048   // 4*H
#define NWG 32    // workgroups cooperating on one LSTM recurrence
#define NT 512    // threads per recurrence workgroup (16 waves)
#define SL 16     // h-columns owned per workgroup (512/32)

typedef __attribute__((ext_vector_type(16))) _Float16 v16h;
typedef __attribute__((ext_vector_type(8)))  _Float16 v8h;
typedef __attribute__((ext_vector_type(8)))  float    v8f;

#define WMMA16x16x32(a, b, c) \
  __builtin_amdgcn_wmma_f32_16x16x32_f16(false, (a), false, (b), (short)0, (c), false, false)

__device__ __forceinline__ v16h frag_cat(v8h lo, v8h hi) {
  v16h r;
#pragma unroll
  for (int j = 0; j < 8; ++j) { r[j] = lo[j]; r[j + 8] = hi[j]; }
  return r;
}

// A fragment (16x32 f16) from a row-major matrix at (row0, k0).
__device__ __forceinline__ v16h load_a_g(const _Float16* __restrict__ A, int ld,
                                         int row0, int k0, int lane) {
  const _Float16* p = A + (size_t)(row0 + (lane & 15)) * ld + (k0 + ((lane >> 4) << 4));
  return frag_cat(*(const v8h*)p, *(const v8h*)(p + 8));
}
// B fragment (32x16) from B^T stored row-major [N, ld] (rows = N).
__device__ __forceinline__ v16h load_bt_g(const _Float16* __restrict__ W, int ld,
                                          int n0, int k0, int lane) {
  const _Float16* p = W + (size_t)(n0 + (lane & 15)) * ld + (k0 + ((lane >> 4) << 4));
  return frag_cat(*(const v8h*)p, *(const v8h*)(p + 8));
}

__device__ __forceinline__ float fsig(float x) { return 1.0f / (1.0f + __expf(-x)); }
__device__ __forceinline__ float ftanh_(float x) {
  float e = __expf(2.0f * x);
  return (e - 1.0f) / (e + 1.0f);
}

// ---- LDS address (byte offset) of a __shared__ object ----------------------
__device__ __forceinline__ unsigned lds_of(const void* p) {
  return (unsigned)(uintptr_t)p;   // low 32 bits of the LDS-aperture flat addr
}

// ---- CDNA5 async global->LDS (ASYNCcnt) — fallback staging path ------------
__device__ __forceinline__ void async_b128(unsigned lds, const void* g) {
  asm volatile("global_load_async_to_lds_b128 %0, %1, off"
               :: "v"(lds), "v"(g) : "memory");
}
__device__ __forceinline__ void wait_async0() {
#if __has_builtin(__builtin_amdgcn_s_wait_asynccnt)
  __builtin_amdgcn_s_wait_asynccnt(0);
#else
  asm volatile("s_wait_asynccnt 0" ::: "memory");
#endif
}

// ---- CDNA5 Tensor Data Mover (TENSORcnt) -----------------------------------
#if __has_builtin(__builtin_amdgcn_tensor_load_to_lds)
#define HAVE_TDM 1
typedef unsigned int u32x4 __attribute__((ext_vector_type(4)));
typedef int          i32x4 __attribute__((ext_vector_type(4)));
typedef int          i32x8 __attribute__((ext_vector_type(8)));

// 2D/3D tiled global->LDS DMA with hardware LDS padding.
//   dim0  : contiguous elements per row (f16), == tensor_dim0_stride
//   rows  : tile_dim1 (== tensor_dim1)
//   tile2 : 0 for 2D, else number of z-blocks (tensor_dim1_stride hops apart)
//   padc  : pad_interval code (6 = every 128 dwords, 7 = every 256 dwords);
//           pad_amount fixed at 4 dwords (16B) -> keeps 16B alignment and
//           shifts consecutive rows across LDS banks.
__device__ __forceinline__ void tdm_load(unsigned lds, const void* g,
                                         unsigned dim0, unsigned rows,
                                         unsigned tile2,
                                         unsigned long long zstride,
                                         unsigned padc) {
  unsigned long long ga = (unsigned long long)(uintptr_t)g;
  u32x4 g0;
  g0[0] = 1u;                                             // count=1 (valid)
  g0[1] = lds;                                            // lds_addr (bytes)
  g0[2] = (unsigned)(ga & 0xFFFFFFFFu);                   // global_addr lo
  g0[3] = (unsigned)((ga >> 32) & 0x1FFFFFFu) | (2u << 30); // addr hi | type=2
  i32x8 g1;
  g1[0] = (int)((1u << 16)      // data_size = 2 bytes
              | (1u << 20)      // pad_enable
              | (padc << 22)    // pad_interval
              | (3u << 25));    // pad_amount = 4 dwords
  g1[1] = (int)((dim0 & 0xFFFFu) << 16);                  // tensor_dim0 lo16
  g1[2] = (int)((dim0 >> 16) | ((rows & 0xFFFFu) << 16)); // dim0 hi | dim1 lo
  g1[3] = (int)((rows >> 16) | ((dim0 & 0xFFFFu) << 16)); // dim1 hi | tile_dim0
  g1[4] = (int)((rows & 0xFFFFu) | (tile2 << 16));        // tile_dim1|tile_dim2
  g1[5] = (int)dim0;                                      // dim0_stride lo32
  g1[6] = (int)((unsigned)(zstride & 0xFFFFu) << 16);     // |dim1_stride lo16
  g1[7] = (int)(unsigned)(zstride >> 16);                 // dim1_stride hi32
  i32x4 g2 = {};
  g2[0] = (int)tile2;                                     // tensor_dim2
  i32x4 g3 = {};
#if __clang_major__ >= 23
  i32x8 z8 = {};
  __builtin_amdgcn_tensor_load_to_lds(g0, g1, g2, g3, z8, 0);
#else
  __builtin_amdgcn_tensor_load_to_lds(g0, g1, g2, g3, 0);
#endif
}
__device__ __forceinline__ void wait_tensor0() {
  __builtin_amdgcn_s_wait_tensorcnt(0);
}
#else
#define HAVE_TDM 0
#endif

// Grid-wide (32-WG) step barrier: monotonic counter, reset by prep each launch.
__device__ __forceinline__ void grid_barrier(unsigned* cnt, unsigned target) {
  __threadfence();
  __syncthreads();
  if (threadIdx.x == 0) {
    __hip_atomic_fetch_add(cnt, 1u, __ATOMIC_ACQ_REL, __HIP_MEMORY_SCOPE_AGENT);
    while (__hip_atomic_load(cnt, __ATOMIC_ACQUIRE, __HIP_MEMORY_SCOPE_AGENT) < target)
      __builtin_amdgcn_s_sleep(2);
  }
  __syncthreads();
  __threadfence();
}

// ---------------------------------------------------------------------------
// Prep: f32->f16 conversions, combined biases, initial h into hbuf slot 0,
// and barrier-counter reset (required every launch for graph replay).
// ---------------------------------------------------------------------------
__global__ void prep_kernel(const float* __restrict__ inputs,
                            const float* __restrict__ W_ih1, const float* __restrict__ W_hh1,
                            const float* __restrict__ b_ih1, const float* __restrict__ b_hh1,
                            const float* __restrict__ W_ih2, const float* __restrict__ W_hh2,
                            const float* __restrict__ b_ih2, const float* __restrict__ b_hh2,
                            const float* __restrict__ W_lin,
                            const float* __restrict__ h1_0, const float* __restrict__ h2_0,
                            _Float16* __restrict__ x16, _Float16* __restrict__ W1cat,
                            _Float16* __restrict__ Wih2h, _Float16* __restrict__ Whh2h,
                            _Float16* __restrict__ Wlinh,
                            _Float16* __restrict__ hbuf1, _Float16* __restrict__ hbuf2,
                            float* __restrict__ b1, float* __restrict__ b2,
                            unsigned* __restrict__ counters) {
  const size_t gid = (size_t)blockIdx.x * blockDim.x + threadIdx.x;
  const size_t stride = (size_t)gridDim.x * blockDim.x;
  for (size_t i = gid; i < (size_t)T_STEPS * BB * II; i += stride)
    x16[i] = (_Float16)inputs[i];
  for (size_t i = gid; i < (size_t)G4 * (II + HH); i += stride) {
    int g = (int)(i / (II + HH)), k = (int)(i % (II + HH));
    float v = (k < II) ? W_ih1[(size_t)g * II + k] : W_hh1[(size_t)g * HH + (k - II)];
    W1cat[i] = (_Float16)v;
  }
  for (size_t i = gid; i < (size_t)G4 * HH; i += stride) {
    Wih2h[i] = (_Float16)W_ih2[i];
    Whh2h[i] = (_Float16)W_hh2[i];
  }
  for (size_t i = gid; i < (size_t)OO * HH; i += stride)
    Wlinh[i] = (_Float16)W_lin[i];
  for (size_t i = gid; i < (size_t)BB * HH; i += stride) {
    hbuf1[i] = (_Float16)h1_0[i];
    hbuf2[i] = (_Float16)h2_0[i];
  }
  for (size_t i = gid; i < (size_t)G4; i += stride) {
    b1[i] = b_ih1[i] + b_hh1[i];
    b2[i] = b_ih2[i] + b_hh2[i];
  }
  if (gid == 0) { counters[0] = 0; counters[1] = 0; }
}

// ---------------------------------------------------------------------------
// Cooperative LSTM recurrence: NWG persistent workgroups (one per WGP).
// WG `wg` owns h-columns [wg*16, wg*16+16) and the 64 gate rows
// g = q*512 + wg*16 + i (q=0..3). The weight slice is TDM-staged into LDS
// ONCE (3D descriptor: NK x 16 rows x 4 gate blocks, HW-padded rows) ->
// zero weight traffic in the loop. Per step: one TDM op refreshes the full h
// matrix from the double-buffered global hbuf, pure LDS->WMMA gate GEMM
// (16 waves x one 16x16 tile), i/f/g/o exchange via LDS gate scratch,
// pointwise c/h update, h-slice store, 1 grid barrier. x_t (encoder) is
// TDM-prefetched one step ahead into a double buffer.
// ---------------------------------------------------------------------------
template <int KX>
__global__ void __launch_bounds__(NT, 1)
lstm_seq(const _Float16* __restrict__ x16,      // (T*64, KX) or nullptr
         const _Float16* __restrict__ Wcat,     // (2048, KX+512) row-major
         const float* __restrict__ bias,        // (2048) or nullptr  (encoder)
         const float* __restrict__ xp_const,    // (64,2048) f32 or nullptr (decoder)
         const float* __restrict__ c0,
         _Float16* __restrict__ hbuf,           // (2, 64, 512) f16; slot0 = h0
         _Float16* __restrict__ hs_out,         // (T*64, 512) f16 or nullptr
         unsigned* __restrict__ counter) {
  constexpr int NK    = KX + HH;
  // LDS padding: 4 dwords (8 halfs) inserted every 2^SEGSH halfs by TDM.
  constexpr int SEGSH = (KX > 0) ? 8 : 9;            // 256-half / 512-half segs
  constexpr unsigned PADC = (KX > 0) ? 6u : 7u;      // 128 / 256 dword interval
  constexpr int WROW  = NK + ((NK >> SEGSH) << 3);   // 792 (enc) / 520 (dec)
  constexpr int HLD   = HH + 8;                      // 520
  constexpr int KXP   = (KX > 0) ? (KX + 8) : 8;     // 264 (enc)
  __shared__ __align__(16) _Float16 Wsl[64 * WROW];  // weight slice
  __shared__ __align__(16) _Float16 Hb[BB][HLD];     // full h, refreshed per step
  __shared__ __align__(16) _Float16 Xb[2][BB][KXP];  // x_t double buffer
  __shared__ float gsc[BB][68];                      // gate scratch (64 cols)
  __shared__ float cst[BB][SL];                      // cell state slice

  const int tid = threadIdx.x, lane = tid & 31, wave = tid >> 5;
  const int lm = lane & 15, lh8 = (lane >> 4) << 3, lh16 = (lane >> 4) << 4;
  const int m0 = (wave >> 2) << 4;                   // 0,16,32,48
  const int nt = wave & 3;                           // gate q of this wave's tile
  const int wg = blockIdx.x;
  const int colbase = wg * SL;

  // ---- one-time staging: weight slice + cell slice + x_0 -------------------
  {
#if HAVE_TDM
    if (wave == 0) {
      // 3D: 4 gate blocks of 16 rows x NK, blocks 512 rows apart in Wcat.
      tdm_load(lds_of(Wsl), Wcat + (size_t)colbase * NK,
               (unsigned)NK, 16u, 4u, (unsigned long long)HH * NK, PADC);
      if (KX > 0)
        tdm_load(lds_of(&Xb[0][0][0]), x16, (unsigned)KX, (unsigned)BB, 0u,
                 (unsigned long long)KX * BB, PADC);
      wait_tensor0();
    }
#else
    constexpr int CPR = NK >> 3;
    for (int c = tid; c < 64 * CPR; c += NT) {
      int lr = c / CPR, col = (c % CPR) << 3;
      int g = ((lr >> 4) * HH) + colbase + (lr & 15);
      int lcol = col + ((col >> SEGSH) << 3);
      async_b128(lds_of(&Wsl[lr * WROW + lcol]), Wcat + (size_t)g * NK + col);
    }
    if (KX > 0) {
      for (int c = tid; c < BB * (KX >> 3); c += NT) {
        int row = c / (KX >> 3), col = (c % (KX >> 3)) << 3;
        async_b128(lds_of(&Xb[0][row][col]), x16 + (size_t)row * KX + col);
      }
    }
    wait_async0();
#endif
    for (int e = tid; e < BB * SL; e += NT)
      cst[e >> 4][e & 15] = c0[(size_t)(e >> 4) * HH + colbase + (e & 15)];
    __syncthreads();
  }

  for (int t = 0; t < T_STEPS; ++t) {
    // ---- stage full h for this step from hbuf[t&1] (written last step) ----
    {
      const _Float16* hsrc = hbuf + (size_t)(t & 1) * BB * HH;
#if HAVE_TDM
      if (wave == 0) {
        tdm_load(lds_of(&Hb[0][0]), hsrc, (unsigned)HH, (unsigned)BB, 0u,
                 (unsigned long long)HH * BB, 7u);
        wait_tensor0();                // also covers x_t issued last iteration
      }
#else
      for (int c = tid; c < BB * (HH >> 3); c += NT) {
        int row = c >> 6, col = (c & 63) << 3;
        async_b128(lds_of(&Hb[row][col]), hsrc + (size_t)row * HH + col);
      }
      wait_async0();
#endif
      __syncthreads();
    }
    // ---- prefetch x_{t+1} while we compute (encoder only) -----------------
    if (KX > 0 && t + 1 < T_STEPS) {
      const _Float16* xsrc = x16 + (size_t)(t + 1) * BB * KX;
      int b = (t + 1) & 1;
#if HAVE_TDM
      if (wave == 0)
        tdm_load(lds_of(&Xb[b][0][0]), xsrc, (unsigned)KX, (unsigned)BB, 0u,
                 (unsigned long long)KX * BB, PADC);
#else
      for (int c = tid; c < BB * (KX >> 3); c += NT) {
        int row = c / (KX >> 3), col = (c % (KX >> 3)) << 3;
        async_b128(lds_of(&Xb[b][row][col]), xsrc + (size_t)row * KX + col);
      }
#endif
    }
    // ---- gate GEMM: one 16x16 tile per wave, all operands in LDS ----------
    {
      const int brow = ((nt << 4) + lm) * WROW;
      v8f a = {};
      if (KX > 0) {
        const _Float16 (*Xc)[KXP] = Xb[t & 1];
#pragma unroll
        for (int k0 = 0; k0 < KX; k0 += 32) {
          const _Float16* ap = &Xc[m0 + lm][k0 + lh16];
          v16h af = frag_cat(*(const v8h*)ap, *(const v8h*)(ap + 8));
          int kk = k0 + lh16;
          const _Float16* bp = &Wsl[brow + kk + ((kk >> SEGSH) << 3)];
          v16h bf = frag_cat(*(const v8h*)bp, *(const v8h*)(bp + 8));
          a = WMMA16x16x32(af, bf, a);
        }
      }
#pragma unroll 8
      for (int k0 = 0; k0 < HH; k0 += 32) {
        const _Float16* ap = &Hb[m0 + lm][k0 + lh16];
        v16h af = frag_cat(*(const v8h*)ap, *(const v8h*)(ap + 8));
        int kk = KX + k0 + lh16;
        const _Float16* bp = &Wsl[brow + kk + ((kk >> SEGSH) << 3)];
        v16h bf = frag_cat(*(const v8h*)bp, *(const v8h*)(bp + 8));
        a = WMMA16x16x32(af, bf, a);
      }
#pragma unroll
      for (int r = 0; r < 8; ++r)
        gsc[m0 + r + lh8][(nt << 4) + lm] = a[r];
    }
    __syncthreads();
    // ---- pointwise LSTM update for our 64x16 slice ------------------------
    {
      _Float16* hdst = hbuf + (size_t)((t + 1) & 1) * BB * HH;
      for (int e = tid; e < BB * SL; e += NT) {
        int m = e >> 4, hc = e & 15;
        float gi = gsc[m][hc], gf = gsc[m][16 + hc];
        float gg = gsc[m][32 + hc], go = gsc[m][48 + hc];
        if (bias) {
          gi += bias[colbase + hc];            gf += bias[HH + colbase + hc];
          gg += bias[2 * HH + colbase + hc];   go += bias[3 * HH + colbase + hc];
        }
        if (xp_const) {
          const float* xp = xp_const + (size_t)m * G4 + colbase + hc;
          gi += xp[0]; gf += xp[HH]; gg += xp[2 * HH]; go += xp[3 * HH];
        }
        float cn = fsig(gf) * cst[m][hc] + fsig(gi) * ftanh_(gg);
        float hn = fsig(go) * ftanh_(cn);
        cst[m][hc] = cn;
        _Float16 h16 = (_Float16)hn;
        hdst[(size_t)m * HH + colbase + hc] = h16;
        if (hs_out) hs_out[((size_t)t * BB + m) * HH + colbase + hc] = h16;
      }
    }
    grid_barrier(counter, (unsigned)(NWG * (t + 1)));
  }
}

// ---------------------------------------------------------------------------
// xp2_const(64,2048) = h1T(64,512) @ W_ih2^T + b2   (wave-per-tile WMMA GEMM)
// ---------------------------------------------------------------------------
__global__ void __launch_bounds__(256)
gemm_xp2(const _Float16* __restrict__ A, const _Float16* __restrict__ W,
         const float* __restrict__ b2, float* __restrict__ out) {
  const int wid = blockIdx.x * 8 + (threadIdx.x >> 5);
  const int lane = threadIdx.x & 31;
  const int m0 = (wid >> 7) << 4, n0 = (wid & 127) << 4;
  v8f acc = {};
  for (int k0 = 0; k0 < HH; k0 += 32) {
    v16h af = load_a_g(A, HH, m0, k0, lane);
    v16h bf = load_bt_g(W, HH, n0, k0, lane);
    acc = WMMA16x16x32(af, bf, acc);
  }
  const int n = n0 + (lane & 15);
  const float bb = b2[n];
#pragma unroll
  for (int r = 0; r < 8; ++r)
    out[(size_t)(m0 + r + ((lane >> 4) << 3)) * G4 + n] = acc[r] + bb;
}

// ---------------------------------------------------------------------------
// out(32768,256) = hs2(32768,512) @ W_lin^T + b_lin  (wave-per-tile WMMA GEMM)
// ---------------------------------------------------------------------------
__global__ void __launch_bounds__(256)
gemm_out(const _Float16* __restrict__ A, const _Float16* __restrict__ W,
         const float* __restrict__ bl, float* __restrict__ out) {
  const int wid = blockIdx.x * 8 + (threadIdx.x >> 5);
  const int lane = threadIdx.x & 31;
  const int m0 = (wid >> 4) << 4, n0 = (wid & 15) << 4;
  v8f acc = {};
  for (int k0 = 0; k0 < HH; k0 += 32) {
    v16h af = load_a_g(A, HH, m0, k0, lane);
    v16h bf = load_bt_g(W, HH, n0, k0, lane);
    acc = WMMA16x16x32(af, bf, acc);
  }
  const int n = n0 + (lane & 15);
  const float bb = bl[n];
#pragma unroll
  for (int r = 0; r < 8; ++r)
    out[(size_t)(m0 + r + ((lane >> 4) << 3)) * OO + n] = acc[r] + bb;
}

// ---------------------------------------------------------------------------
extern "C" void kernel_launch(void* const* d_in, const int* in_sizes, int n_in,
                              void* d_out, int out_size, void* d_ws, size_t ws_size,
                              hipStream_t stream) {
  (void)in_sizes; (void)n_in; (void)out_size; (void)ws_size;
  const float* inputs = (const float*)d_in[0];
  const float* W_ih1  = (const float*)d_in[1];
  const float* W_hh1  = (const float*)d_in[2];
  const float* b_ih1  = (const float*)d_in[3];
  const float* b_hh1  = (const float*)d_in[4];
  const float* W_ih2  = (const float*)d_in[5];
  const float* W_hh2  = (const float*)d_in[6];
  const float* b_ih2  = (const float*)d_in[7];
  const float* b_hh2  = (const float*)d_in[8];
  const float* W_lin  = (const float*)d_in[9];
  const float* b_lin  = (const float*)d_in[10];
  const float* h1_0   = (const float*)d_in[11];
  const float* c1_0   = (const float*)d_in[12];
  const float* h2_0   = (const float*)d_in[13];
  const float* c2_0   = (const float*)d_in[14];
  float* out = (float*)d_out;

  // Workspace layout (~57 MB total, 256B-aligned regions).
  char* ws = (char*)d_ws;
  size_t off = 0;
  auto carve = [&](size_t bytes) {
    char* p = ws + off;
    off += (bytes + 255) & ~(size_t)255;
    return p;
  };
  _Float16* x16   = (_Float16*)carve((size_t)T_STEPS * BB * II * 2); // 16 MB
  _Float16* W1cat = (_Float16*)carve((size_t)G4 * (II + HH) * 2);    // 3 MB
  _Float16* Wih2h = (_Float16*)carve((size_t)G4 * HH * 2);           // 2 MB
  _Float16* Whh2h = (_Float16*)carve((size_t)G4 * HH * 2);           // 2 MB
  _Float16* Wlinh = (_Float16*)carve((size_t)OO * HH * 2);           // 256 KB
  _Float16* hs2   = (_Float16*)carve((size_t)T_STEPS * BB * HH * 2); // 32 MB
  _Float16* hbuf1 = (_Float16*)carve((size_t)2 * BB * HH * 2);       // 128 KB
  _Float16* hbuf2 = (_Float16*)carve((size_t)2 * BB * HH * 2);       // 128 KB
  float*    b1    = (float*)carve((size_t)G4 * 4);
  float*    b2    = (float*)carve((size_t)G4 * 4);
  float*    xp2c  = (float*)carve((size_t)BB * G4 * 4);              // 512 KB
  unsigned* cnts  = (unsigned*)carve(256);

  prep_kernel<<<2048, 256, 0, stream>>>(inputs, W_ih1, W_hh1, b_ih1, b_hh1,
                                        W_ih2, W_hh2, b_ih2, b_hh2, W_lin,
                                        h1_0, h2_0, x16, W1cat, Wih2h, Whh2h,
                                        Wlinh, hbuf1, hbuf2, b1, b2, cnts);
  // Encoder: gates = [x_t | h] @ W1cat^T + (b_ih1+b_hh1); final h -> hbuf1[0].
  lstm_seq<II><<<NWG, NT, 0, stream>>>(x16, W1cat, b1, nullptr, c1_0,
                                       hbuf1, nullptr, cnts + 0);
  // Constant decoder input projection: xp2 = h1T @ W_ih2^T + (b_ih2+b_hh2).
  // T=512 even -> final encoder h landed in hbuf1 slot 0.
  gemm_xp2<<<64, 256, 0, stream>>>(hbuf1, Wih2h, b2, xp2c);
  // Decoder: gates = xp2 + h @ W_hh2^T; every h_t stored (f16) for the linear.
  lstm_seq<0><<<NWG, NT, 0, stream>>>(nullptr, Whh2h, nullptr, xp2c, c2_0,
                                      hbuf2, hs2, cnts + 1);
  // Output linear.
  gemm_out<<<4096, 256, 0, stream>>>(hs2, Wlinh, b_lin, out);
}
